// AdaptiveRotatedConv2d_25847113188003
// MI455X (gfx1250) — matrix operand order repaired and checked by reading, not compiled, not run on hardware
//
#include <hip/hip_runtime.h>

typedef __attribute__((ext_vector_type(8)))  __bf16 v8bf;
typedef __attribute__((ext_vector_type(16))) __bf16 v16bf;
typedef __attribute__((ext_vector_type(8)))  float  v8f;

#define BSZ  16
#define NKER 4
#define CO   256
#define CI   256
#define HH   64
#define WW   64

// LDS pixel stride (bf16 elems): 40 = 20 dwords -> conflict-free half-wave b128 reads
#define PSTR 40

// ---------------------------------------------------------------------------
// Kernel 0: build scaled rotation matrices rm[b][p][n*9+q]  (fp32, 16*9*36)
// ---------------------------------------------------------------------------
__global__ void build_rm_kernel(const float* __restrict__ alphas,
                                const float* __restrict__ angles,
                                float* __restrict__ rm) {
  int t = threadIdx.x;
  if (t >= BSZ * NKER) return;
  float th = angles[t];
  float al = alphas[t];
  float xc = cosf(th), ys = sinf(th);
  float a_ = xc - ys, b_ = xc * ys, c_ = xc + ys;
  int b = t / NKER, n = t % NKER;

  float M[81];
#pragma unroll
  for (int i = 0; i < 81; ++i) M[i] = 0.f;

  if (th >= 0.f) {
    M[0]  = a_;        M[1]  = 1.f - a_;
    M[10] = xc - b_;   M[11] = b_;        M[13] = 1.f - c_ + b_; M[14] = ys - b_;
    M[20] = a_;        M[23] = 1.f - a_;
    M[27] = b_;        M[28] = ys - b_;   M[30] = xc - b_;       M[31] = 1.f - c_ + b_;
    M[40] = 1.f;
    M[49] = 1.f - c_ + b_; M[50] = xc - b_; M[52] = ys - b_;     M[53] = b_;
    M[57] = 1.f - a_;  M[60] = a_;
    M[66] = ys - b_;   M[67] = 1.f - c_ + b_; M[69] = b_;        M[70] = xc - b_;
    M[79] = 1.f - a_;  M[80] = a_;
  } else {
    M[0]  = c_;        M[3]  = 1.f - c_;
    M[9]  = -b_;       M[10] = xc + b_;   M[12] = b_ - ys;       M[13] = 1.f - a_ - b_;
    M[19] = 1.f - c_;  M[20] = c_;
    M[30] = xc + b_;   M[31] = 1.f - a_ - b_; M[33] = -b_;       M[34] = b_ - ys;
    M[40] = 1.f;
    M[46] = b_ - ys;   M[47] = -b_;       M[49] = 1.f - a_ - b_; M[50] = xc + b_;
    M[60] = c_;        M[61] = 1.f - c_;
    M[67] = 1.f - a_ - b_; M[68] = b_ - ys; M[70] = xc + b_;     M[71] = -b_;
    M[77] = 1.f - c_;  M[80] = c_;
  }

  for (int p = 0; p < 9; ++p)
    for (int q = 0; q < 9; ++q)
      rm[(b * 9 + p) * 36 + n * 9 + q] = al * M[p * 9 + q];
}

// ---------------------------------------------------------------------------
// Kernel 1: rotate weights.  rw[b][p][cout][cin] (bf16) =
//           sum_{n,q} rm[b][p][n*9+q] * weight[n][cout][cin][q]
// ---------------------------------------------------------------------------
__global__ __launch_bounds__(256)
void rotate_weights_kernel(const float* __restrict__ weight,
                           const float* __restrict__ rm,
                           __bf16* __restrict__ rw) {
  __shared__ float sRM[9 * 36];
  const int b    = blockIdx.x >> 8;
  const int cout = blockIdx.x & 255;
  const int cin  = threadIdx.x;

  for (int i = threadIdx.x; i < 9 * 36; i += blockDim.x)
    sRM[i] = rm[b * 9 * 36 + i];
  __syncthreads();

  float wv[36];
#pragma unroll
  for (int n = 0; n < NKER; ++n) {
    const float* wp = weight + (((size_t)n * CO + cout) * CI + cin) * 9;
#pragma unroll
    for (int q = 0; q < 9; ++q) wv[n * 9 + q] = wp[q];
  }

#pragma unroll
  for (int p = 0; p < 9; ++p) {
    float acc = 0.f;
#pragma unroll
    for (int j = 0; j < 36; ++j) acc += sRM[p * 36 + j] * wv[j];
    rw[(((size_t)b * 9 + p) * CO + cout) * CI + cin] = (__bf16)acc;
  }
}

// ---------------------------------------------------------------------------
// Kernel 2: per-sample 3x3 conv as 9 shifted bf16 GEMM accumulations.
// Block = (4-row y tile, 64-cout tile, batch); 256 threads = 8 waves.
// Wave (mw,nw): couts [cout0+16*mw,+16) x 2 rows (y0+2*nw..+1) x 64 cols
//   -> 8 accumulator tiles of 16x16 f32, 72 WMMAs per 32-cin chunk.
// ---------------------------------------------------------------------------
union AFrag { v16bf v; v8bf h[2]; };

__global__ __launch_bounds__(256)
void arconv_wmma_kernel(const float* __restrict__ x,
                        const __bf16* __restrict__ rw,
                        float* __restrict__ out) {
  // staging tile: [6 rows][66 cols(halo)][32 cin], pixel stride padded to 40
  __shared__ __align__(64) __bf16 sIn[6 * 66 * PSTR];

  const int y0    = blockIdx.x * 4;   // first output row of tile
  const int ct    = blockIdx.y;       // cout tile 0..3
  const int b     = blockIdx.z;       // batch 0..15
  const int cout0 = ct * 64;
  const int tid   = threadIdx.x;
  const int wid   = tid >> 5;
  const int lane  = tid & 31;
  const int mrow  = lane & 15;
  const int khalf = lane >> 4;        // 0/1
  const int mw    = wid & 3;          // cout sub-tile
  const int nw    = wid >> 2;         // row pair 0/1

  v8f acc[8] = {};                    // 8 n-tiles (2 rows x 4 col-tiles)

  const float* xb = x + (size_t)b * CI * HH * WW;

  for (int c0 = 0; c0 < CI; c0 += 32) {
    // ---- stage 32-cin slab of x (6 rows, halo cols) into LDS as bf16 ----
    for (int e = tid; e < 6 * 66 * 32; e += 256) {
      int xcol = e % 66;
      int cr   = e / 66;              // 0..191
      int cin  = cr & 31;
      int r    = cr >> 5;             // 0..5
      int gy   = y0 + r - 1;
      int gx   = xcol - 1;
      float v  = 0.f;
      if ((unsigned)gy < HH && (unsigned)gx < WW)
        v = xb[((size_t)(c0 + cin) * HH + gy) * WW + gx];
      sIn[(r * 66 + xcol) * PSTR + cin] = (__bf16)v;
    }
    __syncthreads();

#pragma unroll
    for (int p = 0; p < 9; ++p) {
      const int dy = p / 3;           // row offset within staged tile
      const int dx = p % 3;           // col shift (image dx-1 + halo 1)

      // A fragment: rw[b][p][cout0 + 16*mw + mrow][c0 + k]
      // 16-bit A layout: lanes 0-15 hold K {0..7,16..23}; lanes 16-31 {8..15,24..31}
      const __bf16* ap =
          rw + (((size_t)(b * 9 + p) * CO + cout0 + mw * 16 + mrow) * CI
                + c0 + khalf * 8);
      AFrag a;
      a.h[0] = *(const v8bf*)(ap);
      a.h[1] = *(const v8bf*)(ap + 16);

#pragma unroll
      for (int nt = 0; nt < 8; ++nt) {
        const int r    = 2 * nw + (nt >> 2) + dy;       // staged row 0..5
        const int xcol = (nt & 3) * 16 + mrow + dx;     // staged col 0..65
        const __bf16* bp = &sIn[(r * 66 + xcol) * PSTR + khalf * 8];
        AFrag bfr;
        bfr.h[0] = *(const v8bf*)(bp);
        bfr.h[1] = *(const v8bf*)(bp + 16);
        acc[nt] = __builtin_amdgcn_wmma_f32_16x16x32_bf16(
            false, a.v, false, bfr.v, (short)0, acc[nt], false, false);
      }
    }
    __syncthreads();
  }

  // ---- write 64cout x 4row x 64col tile: VGPR rr -> M = rr + 8*khalf ----
  float* ob = out + ((size_t)b * CO + cout0 + mw * 16) * HH * WW;
#pragma unroll
  for (int nt = 0; nt < 8; ++nt) {
    const int row = y0 + 2 * nw + (nt >> 2);
    const int col = (nt & 3) * 16 + mrow;
#pragma unroll
    for (int rr = 0; rr < 8; ++rr) {
      const int m = rr + khalf * 8;
      ob[((size_t)m * HH + row) * WW + col] = acc[nt][rr];
    }
  }
}

// ---------------------------------------------------------------------------
extern "C" void kernel_launch(void* const* d_in, const int* in_sizes, int n_in,
                              void* d_out, int out_size, void* d_ws, size_t ws_size,
                              hipStream_t stream) {
  const float* x      = (const float*)d_in[0];
  const float* alphas = (const float*)d_in[1];
  const float* angles = (const float*)d_in[2];
  const float* weight = (const float*)d_in[3];
  float* out = (float*)d_out;

  float*  rm = (float*)d_ws;                              // 16*9*36 fp32
  __bf16* rw = (__bf16*)((char*)d_ws + 32768);            // 16*9*256*256 bf16

  build_rm_kernel<<<1, 64, 0, stream>>>(alphas, angles, rm);
  rotate_weights_kernel<<<BSZ * CO, 256, 0, stream>>>(weight, rm, rw);
  arconv_wmma_kernel<<<dim3(HH / 4, CO / 64, BSZ), 256, 0, stream>>>(x, rw, out);
}
